// CGCNN_90486370993095
// MI455X (gfx1250) — matrix operand order repaired and checked by reading, not compiled, not run on hardware
//
#include <hip/hip_runtime.h>
#include <hip/hip_bf16.h>

// ---------------------------------------------------------------------------
// CGCNN forward for MI455X (gfx1250, wave32, WMMA).
// Heavy path: two CGConv(128, dim=32) layers -> per-edge [16,288]x[288,128]
// f16 WMMA GEMMs with f32 accumulation. Weights live in the 320KB WGP LDS;
// z-tiles are DOUBLE-BUFFERED and staged with GLOBAL_LOAD_ASYNC_TO_LDS_B128
// (ASYNCcnt) so gather overlaps the WMMA pipeline. Gated message epilogue +
// float atomic scatter-add.
// ---------------------------------------------------------------------------

typedef _Float16 v16h __attribute__((ext_vector_type(16)));
typedef _Float16 hv8  __attribute__((ext_vector_type(8)));
typedef float    v8f  __attribute__((ext_vector_type(8)));

#define NN      50000
#define NE      800000
#define ED      32
#define HID     128
#define NG      256
#define ZIN     288            // 2*HID + ED
#define KT      9              // ZIN / 32
#define ZS      296            // padded z-row stride (halves), 16B aligned
#define WSWH    36864          // swizzled weight halves per matrix (9*2*128*16)

// LDS layout (bytes)
#define LDS_WF   0
#define LDS_WS   73728
#define LDS_Z0   147456
#define ZBYTES   9472          // 16*ZS*2
#define LDS_D0   166400
#define LDS_TOT  166528

// ---- CDNA5 async-to-LDS helpers (inline asm: portable across toolchains) --
__device__ __forceinline__ void async_ld16(void* ldsPtr, const void* g) {
  // the LDS address is taken from VDST; generic LDS pointers carry the wave-relative
  // LDS offset in addr[31:0] (ISA 10.2 aperture mapping), so truncation is
  // the correct VDST operand.
  unsigned loff = (unsigned)(size_t)ldsPtr;
  unsigned long long ga = (unsigned long long)(size_t)g;
  asm volatile("global_load_async_to_lds_b128 %0, %1, off"
               :: "v"(loff), "v"(ga) : "memory");
}

__device__ __forceinline__ void wait_async0() {
#if __has_builtin(__builtin_amdgcn_s_wait_asynccnt)
  __builtin_amdgcn_s_wait_asynccnt(0);
#else
  asm volatile("s_wait_asynccnt 0x0" ::: "memory");
#endif
}

// ---------------- small utility kernels ----------------

__global__ void k_zero_f32(float* __restrict__ p, int n) {
  for (int i = blockIdx.x * blockDim.x + threadIdx.x; i < n;
       i += gridDim.x * blockDim.x)
    p[i] = 0.0f;
}

__global__ void k_cvt_f16(const float* __restrict__ in,
                          _Float16* __restrict__ out, int n) {
  for (int i = blockIdx.x * blockDim.x + threadIdx.x; i < n;
       i += gridDim.x * blockDim.x)
    out[i] = (_Float16)in[i];
}

// Build WMMA-B-fragment-friendly swizzled f16 weights.
// out[((kt*2+kh)*128 + c)*16 + j] = W[(kt*32 + kh*16 + j)*128 + c]
__global__ void k_prep_w(const float* __restrict__ Wf,   // (2,288,128)
                         const float* __restrict__ Ws,   // (2,288,128)
                         _Float16* __restrict__ wsw) {   // 4 * WSWH halves
  const int total = 4 * WSWH;
  for (int i = blockIdx.x * blockDim.x + threadIdx.x; i < total;
       i += gridDim.x * blockDim.x) {
    int which = i / WSWH;
    int r     = i - which * WSWH;
    int j  = r & 15;
    int c  = (r >> 4) & 127;
    int kh = (r >> 11) & 1;
    int kt = r >> 12;
    int K  = kt * 32 + kh * 16 + j;
    int layer = which >> 1;
    const float* W = (which & 1) ? Ws : Wf;
    wsw[i] = (_Float16)W[(size_t)layer * ZIN * HID + (size_t)K * HID + c];
  }
}

// ---------------- conv1: CGConv(channels=3, dim=32), scalar ----------------

__global__ void k_conv1(const float* __restrict__ x,      // (NN,3)
                        const int*   __restrict__ ei,     // (2,NE)
                        const float* __restrict__ ea,     // (NE,32)
                        const float* __restrict__ Wf1,    // (38,3)
                        const float* __restrict__ bf1,    // (3,)
                        const float* __restrict__ Ws1,    // (38,3)
                        const float* __restrict__ bs1,    // (3,)
                        float* __restrict__ agg3) {       // (NN,3)
  int e = blockIdx.x * blockDim.x + threadIdx.x;
  if (e >= NE) return;
  int s = ei[e];
  int d = ei[NE + e];
  float fa[3], sa[3];
#pragma unroll
  for (int c = 0; c < 3; ++c) { fa[c] = bf1[c]; sa[c] = bs1[c]; }
#pragma unroll
  for (int k = 0; k < 3; ++k) {
    float xd = x[(size_t)d * 3 + k];
#pragma unroll
    for (int c = 0; c < 3; ++c) {
      fa[c] += xd * Wf1[k * 3 + c];
      sa[c] += xd * Ws1[k * 3 + c];
    }
  }
#pragma unroll
  for (int k = 0; k < 3; ++k) {
    float xs = x[(size_t)s * 3 + k];
#pragma unroll
    for (int c = 0; c < 3; ++c) {
      fa[c] += xs * Wf1[(3 + k) * 3 + c];
      sa[c] += xs * Ws1[(3 + k) * 3 + c];
    }
  }
  for (int j = 0; j < ED; ++j) {
    float ev = ea[(size_t)e * ED + j];
#pragma unroll
    for (int c = 0; c < 3; ++c) {
      fa[c] += ev * Wf1[(6 + j) * 3 + c];
      sa[c] += ev * Ws1[(6 + j) * 3 + c];
    }
  }
#pragma unroll
  for (int c = 0; c < 3; ++c) {
    float sig = 1.0f / (1.0f + __expf(-fa[c]));
    float sp  = (sa[c] > 20.0f) ? sa[c] : __logf(1.0f + __expf(sa[c]));
    __hip_atomic_fetch_add(&agg3[(size_t)d * 3 + c], sig * sp,
                           __ATOMIC_RELAXED, __HIP_MEMORY_SCOPE_AGENT);
  }
}

// node_proj: h = relu((x + agg3) @ Wp + bp), 3 -> 128; also f16 copy
__global__ void k_proj(const float* __restrict__ x,
                       const float* __restrict__ agg3,
                       const float* __restrict__ Wp,     // (3,128)
                       const float* __restrict__ bp,     // (128,)
                       float* __restrict__ h32,
                       _Float16* __restrict__ h16) {
  int n = blockIdx.x;
  int c = threadIdx.x;
  float v0 = x[(size_t)n * 3 + 0] + agg3[(size_t)n * 3 + 0];
  float v1 = x[(size_t)n * 3 + 1] + agg3[(size_t)n * 3 + 1];
  float v2 = x[(size_t)n * 3 + 2] + agg3[(size_t)n * 3 + 2];
  float o = bp[c] + v0 * Wp[c] + v1 * Wp[HID + c] + v2 * Wp[2 * HID + c];
  o = fmaxf(o, 0.0f);
  h32[(size_t)n * HID + c] = o;
  h16[(size_t)n * HID + c] = (_Float16)o;
}

// ---------------- CGConv(128) edge GEMM + gate + scatter (WMMA) ------------
// Block: 256 threads (8 waves). Each wave owns one 16-column N-tile of the
// 128 output channels. Persistent over 16-edge tiles; z-tile double-buffered
// and staged with async-to-LDS so gather overlaps WMMA compute.
__global__ void k_cgconv(const _Float16* __restrict__ h16,
                         const int*      __restrict__ ei,
                         const _Float16* __restrict__ ea16,
                         const _Float16* __restrict__ wswF,
                         const _Float16* __restrict__ wswS,
                         const float*    __restrict__ bf,
                         const float*    __restrict__ bs,
                         float* __restrict__ agg) {
  extern __shared__ char smem[];
  _Float16* sWF = (_Float16*)(smem + LDS_WF);
  _Float16* sWS = (_Float16*)(smem + LDS_WS);

  // Stage both swizzled f16 weight matrices in LDS (amortized over ~50 tiles)
  {
    const uint4* gf = (const uint4*)wswF;
    const uint4* gs = (const uint4*)wswS;
    uint4* lf = (uint4*)sWF;
    uint4* ls = (uint4*)sWS;
    for (int i = threadIdx.x; i < WSWH / 8; i += 256) { lf[i] = gf[i]; ls[i] = gs[i]; }
  }

  const int lane = threadIdx.x & 31;
  const int wave = threadIdx.x >> 5;        // ntile 0..7
  const int r16  = lane & 15;
  const int kh   = lane >> 4;               // K-half select (contiguous k order used here)
  const int col  = wave * 16 + r16;         // output channel of this lane
  const float bfc = bf[col];
  const float bsc = bs[col];

  const int srow = threadIdx.x >> 4;        // staging row 0..15
  const int q    = threadIdx.x & 15;        // staging chunk

  const int numTiles = NE / 16;
  const int stride   = gridDim.x;

  // async-stage one 16-edge z tile into buffer b (base computed by offset
  // arithmetic each time -- no LDS pointer arrays)
  auto stage = [&](int tile, int b) {
    int e = tile * 16 + srow;
    int s = ei[e];
    int d = ei[NE + e];
    char* zrow = smem + LDS_Z0 + b * ZBYTES + srow * (ZS * 2);
    async_ld16(zrow + q * 16,       h16 + (size_t)d * HID + q * 8);
    async_ld16(zrow + 256 + q * 16, h16 + (size_t)s * HID + q * 8);
    if (q < 4)
      async_ld16(zrow + 512 + q * 16, ea16 + (size_t)e * ED + q * 8);
    if (q == 0) ((int*)(smem + LDS_D0 + b * 64))[srow] = d;
  };

  int buf = 0;
  if (blockIdx.x < numTiles) stage(blockIdx.x, 0);
  __syncthreads();   // weights staged (and first-tile ds_store of dst idx)

  for (int tile = blockIdx.x; tile < numTiles; tile += stride) {
    wait_async0();        // this wave's async stage of sZ[buf] complete
    __syncthreads();      // all waves' stages visible; prior reads retired
    int nxt = tile + stride;
    if (nxt < numTiles) stage(nxt, buf ^ 1);   // overlap with compute below

    // ---- 16x128 x2 GEMM: K = 288 in 9 WMMA steps per matrix ----
    v8f accf = {};
    v8f accs = {};
    const _Float16* arow = (const _Float16*)(smem + LDS_Z0 + buf * ZBYTES) +
                           r16 * ZS;
    const int* sDst = (const int*)(smem + LDS_D0 + buf * 64);
#pragma unroll
    for (int kt = 0; kt < KT; ++kt) {
      // A fragment (16x32 f16): lane<16 holds K in [0..15] of k-tile
      hv8 alo = *(const hv8*)(arow + kt * 32 + kh * 16);
      hv8 ahi = *(const hv8*)(arow + kt * 32 + kh * 16 + 8);
      v16h a = __builtin_shufflevector(alo, ahi, 0, 1, 2, 3, 4, 5, 6, 7,
                                       8, 9, 10, 11, 12, 13, 14, 15);
      // B fragments: 32 contiguous bytes in the swizzled layout
      const _Float16* bfp = sWF + ((size_t)(kt * 2 + kh) * HID + col) * 16;
      hv8 bf0  = *(const hv8*)(bfp);
      hv8 bf1v = *(const hv8*)(bfp + 8);
      v16h bF = __builtin_shufflevector(bf0, bf1v, 0, 1, 2, 3, 4, 5, 6, 7,
                                        8, 9, 10, 11, 12, 13, 14, 15);
      accf = __builtin_amdgcn_wmma_f32_16x16x32_f16(
          false, a, false, bF, (short)0, accf, false, false);

      const _Float16* bsp = sWS + ((size_t)(kt * 2 + kh) * HID + col) * 16;
      hv8 bs0  = *(const hv8*)(bsp);
      hv8 bs1v = *(const hv8*)(bsp + 8);
      v16h bS = __builtin_shufflevector(bs0, bs1v, 0, 1, 2, 3, 4, 5, 6, 7,
                                        8, 9, 10, 11, 12, 13, 14, 15);
      accs = __builtin_amdgcn_wmma_f32_16x16x32_f16(
          false, a, false, bS, (short)0, accs, false, false);
    }

    // ---- gate + scatter: D layout VGPR i, lane l -> row i + 8*(l>=16) ----
#pragma unroll
    for (int i = 0; i < 8; ++i) {
      int row = i + 8 * kh;
      int dn  = sDst[row];
      float f  = accf[i] + bfc;
      float sv = accs[i] + bsc;
      float sig = 1.0f / (1.0f + __expf(-f));
      float sp  = (sv > 20.0f) ? sv : __logf(1.0f + __expf(sv));
      __hip_atomic_fetch_add(&agg[(size_t)dn * HID + col], sig * sp,
                             __ATOMIC_RELAXED, __HIP_MEMORY_SCOPE_AGENT);
    }
    buf ^= 1;
  }
}

// h = relu(h + agg); refresh f16 copy
__global__ void k_relu_add(float* __restrict__ h32,
                           const float* __restrict__ agg,
                           _Float16* __restrict__ h16) {
  for (int i = blockIdx.x * blockDim.x + threadIdx.x; i < NN * HID;
       i += gridDim.x * blockDim.x) {
    float v = fmaxf(h32[i] + agg[i], 0.0f);
    h32[i] = v;
    h16[i] = (_Float16)v;
  }
}

// ---------------- pooling + readout ----------------

__global__ void k_pool(const float* __restrict__ h32,
                       const int* __restrict__ batch,
                       float* __restrict__ sums,
                       float* __restrict__ counts) {
  int n = blockIdx.x;
  int c = threadIdx.x;
  int g = batch[n];
  __hip_atomic_fetch_add(&sums[(size_t)g * HID + c], h32[(size_t)n * HID + c],
                         __ATOMIC_RELAXED, __HIP_MEMORY_SCOPE_AGENT);
  if (c == 0)
    __hip_atomic_fetch_add(&counts[g], 1.0f, __ATOMIC_RELAXED,
                           __HIP_MEMORY_SCOPE_AGENT);
}

__global__ void k_graph_mlp(const float* __restrict__ sums,
                            const float* __restrict__ counts,
                            const float* __restrict__ W1,   // (128,128)
                            const float* __restrict__ b1,
                            float* __restrict__ gact) {
  __shared__ float grow[HID];
  int g = blockIdx.x;
  int c = threadIdx.x;
  float cnt = fmaxf(counts[g], 1.0f);
  grow[c] = sums[(size_t)g * HID + c] / cnt;
  __syncthreads();
  float o = b1[c];
  for (int k = 0; k < HID; ++k) o += grow[k] * W1[k * HID + c];
  gact[(size_t)g * HID + c] = fmaxf(o, 0.0f);
}

__global__ void k_heads(const float* __restrict__ gact,
                        const float* __restrict__ Wh,   // (5,128)
                        const float* __restrict__ bh,   // (5,)
                        float* __restrict__ out) {      // (NG,5)
  int idx = blockIdx.x * blockDim.x + threadIdx.x;
  if (idx >= NG * 5) return;
  int g = idx / 5, h = idx % 5;
  float o = bh[h];
  for (int k = 0; k < HID; ++k) o += gact[(size_t)g * HID + k] * Wh[h * HID + k];
  out[idx] = o;
}

// ---------------- launcher ----------------

extern "C" void kernel_launch(void* const* d_in, const int* in_sizes, int n_in,
                              void* d_out, int out_size, void* d_ws, size_t ws_size,
                              hipStream_t stream) {
  (void)in_sizes; (void)n_in; (void)out_size; (void)ws_size;
  const float* x        = (const float*)d_in[0];
  const int*   ei       = (const int*)d_in[1];
  const float* ea       = (const float*)d_in[2];
  const int*   batch    = (const int*)d_in[3];
  const float* Wf1      = (const float*)d_in[4];
  const float* bf1      = (const float*)d_in[5];
  const float* Ws1      = (const float*)d_in[6];
  const float* bs1      = (const float*)d_in[7];
  const float* Wp       = (const float*)d_in[8];
  const float* bp       = (const float*)d_in[9];
  const float* Wf_convs = (const float*)d_in[10];
  const float* bf_convs = (const float*)d_in[11];
  const float* Ws_convs = (const float*)d_in[12];
  const float* bs_convs = (const float*)d_in[13];
  const float* W1       = (const float*)d_in[14];
  const float* b1       = (const float*)d_in[15];
  const float* Wh       = (const float*)d_in[16];
  const float* bh       = (const float*)d_in[17];
  float* out = (float*)d_out;

  // workspace carve-up (256B aligned)
  char* base = (char*)d_ws;
  size_t off = 0;
  auto alloc = [&](size_t bytes) {
    size_t p = off;
    off = (off + bytes + 255) & ~(size_t)255;
    return p;
  };
  float*    h32   = (float*)(base + alloc((size_t)NN * HID * 4));
  _Float16* h16   = (_Float16*)(base + alloc((size_t)NN * HID * 2));
  float*    agg   = (float*)(base + alloc((size_t)NN * HID * 4));
  float*    agg3  = (float*)(base + alloc((size_t)NN * 3 * 4));
  _Float16* ea16  = (_Float16*)(base + alloc((size_t)NE * ED * 2));
  _Float16* wsw   = (_Float16*)(base + alloc((size_t)4 * WSWH * 2));
  float*    sums  = (float*)(base + alloc((size_t)NG * HID * 4));
  float*    cnts  = (float*)(base + alloc((size_t)NG * 4));
  float*    gact  = (float*)(base + alloc((size_t)NG * HID * 4));

  // prep
  k_zero_f32<<<256, 256, 0, stream>>>(agg3, NN * 3);
  k_zero_f32<<<256, 256, 0, stream>>>(sums, NG * HID);
  k_zero_f32<<<4, 256, 0, stream>>>(cnts, NG);
  k_cvt_f16<<<8192, 256, 0, stream>>>(ea, ea16, NE * ED);
  k_prep_w<<<576, 256, 0, stream>>>(Wf_convs, Ws_convs, wsw);

  // conv1 (3-channel) + projection to 128
  k_conv1<<<(NE + 255) / 256, 256, 0, stream>>>(x, ei, ea, Wf1, bf1, Ws1, bs1, agg3);
  k_proj<<<NN, HID, 0, stream>>>(x, agg3, Wp, bp, h32, h16);

  // two CGConv(128) layers: WMMA edge GEMM + gated scatter
  for (int l = 0; l < 2; ++l) {
    k_zero_f32<<<4096, 256, 0, stream>>>(agg, NN * HID);
    k_cgconv<<<1024, 256, LDS_TOT, stream>>>(
        h16, ei, ea16,
        wsw + (size_t)(l * 2 + 0) * WSWH,
        wsw + (size_t)(l * 2 + 1) * WSWH,
        bf_convs + l * HID, bs_convs + l * HID, agg);
    k_relu_add<<<4096, 256, 0, stream>>>(h32, agg, h16);
  }

  // mean pool + MLP + heads
  k_pool<<<NN, HID, 0, stream>>>(h32, batch, sums, cnts);
  k_graph_mlp<<<NG, HID, 0, stream>>>(sums, cnts, W1, b1, gact);
  k_heads<<<5, 256, 0, stream>>>(gact, Wh, bh, out);
}
